// NoisyTopKGating_85847806312744
// MI455X (gfx1250) — matrix operand order repaired and checked by reading, not compiled, not run on hardware
//
#include <hip/hip_runtime.h>

typedef __attribute__((ext_vector_type(16))) _Float16 v16h;
typedef __attribute__((ext_vector_type(8)))  _Float16 v8h;
typedef __attribute__((ext_vector_type(4)))  _Float16 v4h;
typedef __attribute__((ext_vector_type(8)))  float    v8f;
typedef __attribute__((ext_vector_type(4)))  float    v4f;

#define DDIM   1024          // reduction dim
#define EXP    64            // experts
#define WPITCH 1032          // halves per W row in LDS (1024 + 8 pad)
#define LPITCH 68            // floats per logit row in LDS (64 + 4 pad)
#define NWAVES 8
#define ROWS_PER_BLOCK 128   // 8 waves * 16 rows

__global__ __launch_bounds__(256) void noisy_topk_gating_kernel(
    const float* __restrict__ x, const float* __restrict__ W,
    const float* __restrict__ bias, const float* __restrict__ noise,
    float* __restrict__ out, int N)
{
  extern __shared__ char smem[];
  _Float16* Wl = (_Float16*)smem;                                  // 64 x 1032 halves
  float* Lg = (float*)(smem + (size_t)EXP * WPITCH * sizeof(_Float16));
  float* T2 = Lg + NWAVES * 16 * LPITCH;                           // per row: w0,w1,i0,i1

  const int tid  = threadIdx.x;
  const int wave = tid >> 5;
  const int lane = tid & 31;
  const int m    = lane & 15;
  const int hl   = lane >> 4;

  // ---- stage W (f32 -> f16) into LDS, padded pitch ----
  {
    const v4f* W4 = (const v4f*)W;
    #pragma unroll 4
    for (int i = 0; i < (EXP * DDIM) / 4 / 256; ++i) {
      int f = tid + (i << 8);           // float4 index
      v4f w = W4[f];
      int elem = f << 2;
      int e = elem >> 10;               // expert row
      int d = elem & (DDIM - 1);        // d-offset
      v4h h;
      h.x = (_Float16)w.x; h.y = (_Float16)w.y;
      h.z = (_Float16)w.z; h.w = (_Float16)w.w;
      *(v4h*)(Wl + e * WPITCH + d) = h;
    }
  }
  __syncthreads();

  const int rowBase = blockIdx.x * ROWS_PER_BLOCK + wave * 16;
  const float* xrow = x + (size_t)(rowBase + m) * DDIM;

  // ---- main GEMM loop: 16 rows x 64 experts per wave ----
  v8f acc[4] = {};
  #pragma unroll 2
  for (int kb = 0; kb < DDIM / 32; ++kb) {
    // A fragment (16x32 f16): lane<16 -> K {0..7,16..23}, lane>=16 -> K {8..15,24..31}
    const int ks = kb * 32 + hl * 8;
    v4f a0 = *(const v4f*)(xrow + ks);
    v4f a1 = *(const v4f*)(xrow + ks + 4);
    v4f a2 = *(const v4f*)(xrow + ks + 16);
    v4f a3 = *(const v4f*)(xrow + ks + 20);
    v16h a;
    a[0]=(_Float16)a0.x;  a[1]=(_Float16)a0.y;  a[2]=(_Float16)a0.z;  a[3]=(_Float16)a0.w;
    a[4]=(_Float16)a1.x;  a[5]=(_Float16)a1.y;  a[6]=(_Float16)a1.z;  a[7]=(_Float16)a1.w;
    a[8]=(_Float16)a2.x;  a[9]=(_Float16)a2.y;  a[10]=(_Float16)a2.z; a[11]=(_Float16)a2.w;
    a[12]=(_Float16)a3.x; a[13]=(_Float16)a3.y; a[14]=(_Float16)a3.z; a[15]=(_Float16)a3.w;

    // B fragments (32x16 f16): lane<16 -> K {0..7,16..23}, lane>=16 -> K {8..15,24..31} of one expert column
    #pragma unroll
    for (int nt = 0; nt < 4; ++nt) {
      const _Float16* bp = Wl + (nt * 16 + m) * WPITCH + kb * 32 + hl * 8;
      v8h b0 = *(const v8h*)bp;
      v8h b1 = *(const v8h*)(bp + 16);
      v16h bb;
      #pragma unroll
      for (int j = 0; j < 8; ++j) { bb[j] = b0[j]; bb[j + 8] = b1[j]; }
      acc[nt] = __builtin_amdgcn_wmma_f32_16x16x32_f16(
          false, a, false, bb, (short)0, acc[nt], false, false);
    }
  }

  // ---- epilogue: noisy logits -> wave-private LDS (C layout: VGPR r -> M=r / r+8) ----
  float* lg = Lg + wave * 16 * LPITCH;
  #pragma unroll
  for (int nt = 0; nt < 4; ++nt) {
    const int col = nt * 16 + m;
    const float bv = bias[col];
    #pragma unroll
    for (int r = 0; r < 8; ++r) {
      const int row = hl ? (r + 8) : r;
      float v = acc[nt][r] + bv + noise[(size_t)(rowBase + row) * EXP + col];
      lg[row * LPITCH + col] = v;
    }
  }
  __builtin_amdgcn_wave_barrier();   // DS ops are in-order per wave; fence scheduling

  // ---- top-2 + softmax, one row per lane (lanes 0..15) ----
  if (lane < 16) {
    float b0v = -3.4e38f, b1v = -3.4e38f;
    int i0 = 0, i1 = 0;
    const float* rp = lg + lane * LPITCH;
    #pragma unroll 8
    for (int e = 0; e < EXP; ++e) {
      float v = rp[e];
      if (v > b0v)      { b1v = b0v; i1 = i0; b0v = v; i0 = e; }
      else if (v > b1v) { b1v = v; i1 = e; }
    }
    float dexp = __expf(b1v - b0v);       // b1v <= b0v: stable
    float inv  = 1.0f / (1.0f + dexp);
    float* t = T2 + (wave * 16 + lane) * 4;
    t[0] = inv;                            // weight at i0
    t[1] = dexp * inv;                     // weight at i1
    t[2] = __int_as_float(i0);
    t[3] = __int_as_float(i1);
  }
  __builtin_amdgcn_wave_barrier();

  // ---- scatter-write full 16x64 output tile (coalesced 128B stores) ----
  float* ob = out + (size_t)rowBase * EXP;
  #pragma unroll
  for (int i = 0; i < 32; ++i) {
    int flat = i * 32 + lane;
    int row = flat >> 6;
    int col = flat & 63;
    const float* t = T2 + (wave * 16 + row) * 4;
    float v = 0.0f;
    if (col == __float_as_int(t[2]))      v = t[0];
    else if (col == __float_as_int(t[3])) v = t[1];
    ob[flat] = v;
  }
}

extern "C" void kernel_launch(void* const* d_in, const int* in_sizes, int n_in,
                              void* d_out, int out_size, void* d_ws, size_t ws_size,
                              hipStream_t stream) {
  const float* x     = (const float*)d_in[0];
  const float* W     = (const float*)d_in[1];
  const float* bias  = (const float*)d_in[2];
  const float* noise = (const float*)d_in[3];
  float* out = (float*)d_out;

  const int N = in_sizes[0] / DDIM;            // 131072
  const int blocks = N / ROWS_PER_BLOCK;       // 1024

  const size_t shmem = (size_t)EXP * WPITCH * sizeof(_Float16)     // ~129 KB W
                     + (size_t)NWAVES * 16 * LPITCH * sizeof(float) // ~34 KB logits
                     + (size_t)NWAVES * 16 * 4 * sizeof(float);     // 2 KB top2

  hipFuncSetAttribute((const void*)noisy_topk_gating_kernel,
                      hipFuncAttributeMaxDynamicSharedMemorySize, (int)shmem);
  noisy_topk_gating_kernel<<<blocks, 256, shmem, stream>>>(x, W, bias, noise, out, N);
}
